// ESIM_28003186770376
// MI455X (gfx1250) — matrix operand order repaired and checked by reading, not compiled, not run on hardware
//
#include <hip/hip_runtime.h>

// ============================================================================
// ESIM on gfx1250 (MI455X), bf16 WMMA everywhere, fp32 accumulate.
//
// Padded geometry (all K multiples of 32):
//   H 300 -> Hp 320, 4H 1200 -> 1280, 2H 600 -> D 640, 8H 2400 -> 2560,
//   E 300 -> 320.  T=64 (both P and L), Bt=128 batch.
// Padding is zero-filled and provably stays zero through the LSTM.
//
// WMMA 16x16x32 bf16 fragment layout assumptions (CDNA5 ISA):
//   A (16Mx32K): lane<16: M=lane, halves 0..7 = K k0..k0+7, halves 8..15 =
//                K k0+16..k0+23 ; lanes 16..31: same M mapping, K offset +8.
//   B (32Kx16N): lane<16: N=lane, halves 0..7 = K k0..k0+7, halves 8..15 =
//                K k0+16..k0+23 ; lanes 16..31: K offset +8 (same order as A).
//   C/D (v8f):   element r -> (m = r + 8*(lane>=16), n = lane&15).
//
// New this round: the GEMM stages its LDS tiles with
//   GLOBAL_LOAD_ASYNC_TO_LDS_B128  (ASYNCcnt, VGPR-bypassing DMA path)
// followed by s_wait_asynccnt 0 + workgroup barrier before WMMA consumption.
// ============================================================================

typedef __attribute__((ext_vector_type(16))) __bf16 v16bf;
typedef __attribute__((ext_vector_type(8)))  float  v8f;
typedef __attribute__((ext_vector_type(4)))  unsigned int v4u;

union Frag { v16bf v; v4u q[2]; };

__device__ inline void ldA(Frag& f, const unsigned short* base, int ld,
                           int row0, int k0, int lane) {
  int m = row0 + (lane & 15);
  int k = k0 + ((lane >> 4) << 3);            // +0 or +8
  const unsigned short* p = base + (size_t)m * ld + k;
  f.q[0] = *(const v4u*)p;                     // K k..k+7
  f.q[1] = *(const v4u*)(p + 16);              // K k+16..k+23
}
__device__ inline void ldB(Frag& f, const unsigned short* base, int ld,
                           int col0, int k0, int lane) {
  int n = col0 + (lane & 15);
  int k = k0 + ((lane >> 4) << 3);             // +0 or +8
  const unsigned short* p = base + (size_t)n * ld + k;
  f.q[0] = *(const v4u*)p;                     // K k..k+7
  f.q[1] = *(const v4u*)(p + 16);              // K k+16..k+23
}
__device__ inline v8f wmma_bf(const Frag& a, const Frag& b, v8f c) {
  return __builtin_amdgcn_wmma_f32_16x16x32_bf16(false, a.v, false, b.v,
                                                 (short)0, c, false, false);
}

__device__ inline unsigned short f2bf(float f) {
  unsigned u = __float_as_uint(f);
  unsigned r = u + 0x7FFFu + ((u >> 16) & 1u);
  return (unsigned short)(r >> 16);
}
__device__ inline float bf2f(unsigned short h) {
  return __uint_as_float(((unsigned)h) << 16);
}
__device__ inline float sigm(float x) { return 1.f / (1.f + __expf(-x)); }

// Async global->LDS 16B copy (per active lane). lds_off = low 32 bits of the
// generic address of the LDS destination (aperture maps addr[31:0] to LDS).
__device__ inline void async_cp16(unsigned lds_off, const void* gptr) {
  asm volatile("global_load_async_to_lds_b128 %0, %1, off"
               :: "v"(lds_off), "v"((unsigned long long)(size_t)gptr)
               : "memory");
}
__device__ inline void wait_async0() {
  asm volatile("s_wait_asynccnt 0" ::: "memory");
}

extern __shared__ char smem_c[];

// ---------------------------------------------------------------------------
// Weight prep: fp32 [1200, Kin] -> bf16 padded [1280, Kpad]; bias = bih+bhh.
// Row n = g*320 + j maps to source row g*300 + j (j<300).
// Kin==2400: padded k' = seg*640 + u ; u<320 -> seg*600+u, u>=320 -> +300.
// ---------------------------------------------------------------------------
__global__ void k_prep(const float* Wih, const float* Whh,
                       const float* bih, const float* bhh, int Kin, int Kpad,
                       unsigned short* Wih_o, unsigned short* Whh_o,
                       float* bias_o) {
  int n = blockIdx.x;                  // 0..1279
  int g = n / 320, j = n % 320;
  bool vr = j < 300;
  int sr = g * 300 + j;
  for (int k = threadIdx.x; k < Kpad; k += blockDim.x) {
    float val = 0.f;
    if (vr) {
      int sk; bool vk;
      if (Kin == 300) { sk = k; vk = k < 300; }
      else {
        int seg = k / 640, u = k % 640;
        if (u < 320) { sk = seg * 600 + u;            vk = u < 300; }
        else         { sk = seg * 600 + 300 + (u-320); vk = (u-320) < 300; }
      }
      if (vk) val = Wih[(size_t)sr * Kin + sk];
    }
    Wih_o[(size_t)n * Kpad + k] = f2bf(val);
  }
  for (int k = threadIdx.x; k < 320; k += blockDim.x) {
    float val = (vr && k < 300) ? Whh[(size_t)sr * 300 + k] : 0.f;
    Whh_o[(size_t)n * 320 + k] = f2bf(val);
  }
  if (threadIdx.x == 0) bias_o[n] = vr ? (bih[sr] + bhh[sr]) : 0.f;
}

// ---------------------------------------------------------------------------
// Embedding gather -> bf16 X [T*Bt, 320], pad zeros.
// ---------------------------------------------------------------------------
__global__ void k_embed(const int* tok, const float* emb, unsigned short* X) {
  int i = blockIdx.x;                  // token slot (t*128+b)
  int t = tok[i];
  for (int e = threadIdx.x; e < 320; e += blockDim.x) {
    float v = (e < 300) ? emb[(size_t)t * 300 + e] : 0.f;
    X[(size_t)i * 320 + e] = f2bf(v);
  }
}

// ---------------------------------------------------------------------------
// Batched input GEMM: Z[z] = X[z] @ W[z]^T + bias[z]
//   X: [8192, K] bf16 row-major; W: [1280, K] bf16 (already "B" layout);
//   Z: [8192, 1280] fp32.  Block 256 thr / 8 waves, tile 128x128, BK=32.
//   LDS staged via GLOBAL_LOAD_ASYNC_TO_LDS_B128 (ASYNCcnt).
// ---------------------------------------------------------------------------
struct GemmP {
  const unsigned short* X[4];
  const unsigned short* W[4];
  const float* bias[4];
  float* Z[4];
  int K;
};

__global__ __launch_bounds__(256) void k_gemm(GemmP P) {
  __shared__ unsigned short As[128 * 32];
  __shared__ unsigned short Bs[128 * 32];
  int z = blockIdx.z;
  const unsigned short* X = P.X[z];
  const unsigned short* W = P.W[z];
  int m0 = blockIdx.x * 128, n0 = blockIdx.y * 128;
  int wave = threadIdx.x >> 5, lane = threadIdx.x & 31;
  int row = threadIdx.x >> 1, half = threadIdx.x & 1;

  unsigned ldsA = (unsigned)(size_t)&As[row * 32 + half * 16];
  unsigned ldsB = (unsigned)(size_t)&Bs[row * 32 + half * 16];

  v8f acc[8];
  for (int i = 0; i < 8; i++) acc[i] = v8f{};

  for (int kb = 0; kb < P.K; kb += 32) {
    const unsigned short* gx = X + (size_t)(m0 + row) * P.K + kb + half * 16;
    const unsigned short* gw = W + (size_t)(n0 + row) * P.K + kb + half * 16;
    if (kb + 32 < P.K) {                       // software prefetch next chunk
      __builtin_prefetch(gx + 32, 0, 1);
      __builtin_prefetch(gw + 32, 0, 1);
    }
    // DMA 32B per thread for A and B tiles, bypassing VGPRs.
    async_cp16(ldsA,      gx);
    async_cp16(ldsA + 16, gx + 8);
    async_cp16(ldsB,      gw);
    async_cp16(ldsB + 16, gw + 8);
    wait_async0();
    __syncthreads();
    Frag af; ldA(af, As, 32, wave * 16, 0, lane);
    for (int nt = 0; nt < 8; nt++) {
      Frag bf; ldB(bf, Bs, 32, nt * 16, 0, lane);
      acc[nt] = wmma_bf(af, bf, acc[nt]);
    }
    __syncthreads();
  }
  for (int nt = 0; nt < 8; nt++)
    for (int r = 0; r < 8; r++) {
      int m = m0 + wave * 16 + r + ((lane >> 4) << 3);
      int n = n0 + nt * 16 + (lane & 15);
      P.Z[z][(size_t)m * 1280 + n] = acc[nt][r] + P.bias[z][n];
    }
}

// ---------------------------------------------------------------------------
// Persistent LSTM recurrence. One 32-wave block per chain; h (bf16) lives in
// LDS, c lives in registers (5 C-tiles per wave).  Per step:
//   z = Zin[t] + h @ Whh^T ; gates ; write h to LDS + global.
// ---------------------------------------------------------------------------
struct LstmP {
  const float* Zin[4];
  const unsigned short* Whh[4];     // [1280, 320] bf16
  unsigned short* hs_bf[4];         // [64,128,320] bf16 or null
  float* hs_f[4];                   // [64,128,320] fp32 or null
  int rev[4];
};

__global__ __launch_bounds__(1024) void k_lstm(LstmP P) {
  unsigned short* hsh = (unsigned short*)smem_c;      // 128*320 bf16 = 80KB
  int z = blockIdx.x;
  int lane = threadIdx.x & 31, wave = threadIdx.x >> 5;
  const unsigned short* Whh = P.Whh[z];

  float c[5][8], hn[5][8];
  for (int p = 0; p < 5; p++)
    for (int r = 0; r < 8; r++) c[p][r] = 0.f;
  for (int i = threadIdx.x; i < 128 * 320; i += blockDim.x) hsh[i] = 0;
  __syncthreads();

  for (int tt = 0; tt < 64; tt++) {
    int t = P.rev[z] ? (63 - tt) : tt;
    const float* zt = P.Zin[z] + (size_t)t * 128 * 1280;
    for (int pp = 0; pp < 5; pp++) {
      int pair = wave * 5 + pp;
      int mt = pair / 20, nt = pair % 20;
      v8f a4[4];
      for (int g = 0; g < 4; g++) {
        a4[g] = v8f{};
        for (int r = 0; r < 8; r++) {
          int m = mt * 16 + r + ((lane >> 4) << 3);
          int n = g * 320 + nt * 16 + (lane & 15);
          a4[g][r] = zt[(size_t)m * 1280 + n];
        }
      }
      for (int kc = 0; kc < 10; kc++) {
        int k0 = kc * 32;
        Frag af; ldA(af, hsh, 320, mt * 16, k0, lane);
        for (int g = 0; g < 4; g++) {
          Frag bf; ldB(bf, Whh, 320, g * 320 + nt * 16, k0, lane);
          a4[g] = wmma_bf(af, bf, a4[g]);
        }
      }
      for (int r = 0; r < 8; r++) {
        float ig = sigm(a4[0][r]);
        float fg = sigm(a4[1][r]);
        float gg = tanhf(a4[2][r]);
        float og = sigm(a4[3][r]);
        float cn = fg * c[pp][r] + ig * gg;
        c[pp][r] = cn;
        hn[pp][r] = og * tanhf(cn);
      }
    }
    __syncthreads();   // all reads of h done
    for (int pp = 0; pp < 5; pp++) {
      int pair = wave * 5 + pp;
      int mt = pair / 20, nt = pair % 20;
      for (int r = 0; r < 8; r++) {
        int m = mt * 16 + r + ((lane >> 4) << 3);
        int n = nt * 16 + (lane & 15);
        unsigned short hb = f2bf(hn[pp][r]);
        hsh[m * 320 + n] = hb;
        size_t go = ((size_t)t * 128 + m) * 320 + n;
        if (P.hs_bf[z]) P.hs_bf[z][go] = hb;
        if (P.hs_f[z])  P.hs_f[z][go]  = hn[pp][r];
      }
    }
    __syncthreads();   // h published for next step
  }
}

// ---------------------------------------------------------------------------
// Cross-attention + m_a/m_b construction. One block per batch b (256 thr).
// S = Pr . Hy^T (K=640), Ex = exp(S), masked row/col softmax, then
// prem_attn = alpha @ Hy, hyp_attn = beta^T @ Pr via LDS-transposed B.
// ---------------------------------------------------------------------------
__global__ __launch_bounds__(256)
void k_attn(const int* prem_tok, const int* hyp_tok,
            const unsigned short* pf, const unsigned short* pb,
            const unsigned short* hf, const unsigned short* hb,
            unsigned short* ma, unsigned short* mb_) {
  unsigned short* Tb  = (unsigned short*)smem_c;            // [640][64] 80KB
  float*          Exs = (float*)(smem_c + 81920);           // [64][64] 16KB
  unsigned short* Alp = (unsigned short*)(smem_c + 98304);  // [64][64] 8KB
  unsigned short* Bet = (unsigned short*)(smem_c + 106496); // [64][64] 8KB
  float* rs = (float*)(smem_c + 114688);
  float* cs = (float*)(smem_c + 114944);
  float* mh = (float*)(smem_c + 115200);
  float* mp = (float*)(smem_c + 115456);

  int b = blockIdx.x;
  int tid = threadIdx.x, lane = tid & 31, wave = tid >> 5;

  if (tid < 64)        mh[tid] = (hyp_tok[tid * 128 + b] != 1) ? 1.f : 0.f;
  else if (tid < 128)  mp[tid - 64] = (prem_tok[(tid - 64) * 128 + b] != 1) ? 1.f : 0.f;

  // HyT[d][h]
  for (int i = tid; i < 640 * 64; i += 256) {
    int h = i / 640, d = i % 640;
    unsigned short v = (d < 320) ? hf[((size_t)h * 128 + b) * 320 + d]
                                 : hb[((size_t)h * 128 + b) * 320 + d - 320];
    Tb[d * 64 + h] = v;
  }
  __syncthreads();

  // S -> Ex (A: prem rows from global, B: hyp rows from global)
  for (int ti = wave; ti < 16; ti += 8) {
    int pt = ti >> 2, ht = ti & 3;
    v8f cc = v8f{};
    for (int kc = 0; kc < 20; kc++) {
      int k0 = kc * 32;
      Frag af;
      {
        int p = pt * 16 + (lane & 15);
        int k = k0 + ((lane >> 4) << 3);
        const unsigned short* src = (k < 320)
            ? pf + ((size_t)p * 128 + b) * 320 + k
            : pb + ((size_t)p * 128 + b) * 320 + (k - 320);
        af.q[0] = *(const v4u*)src;
        af.q[1] = *(const v4u*)(src + 16);
      }
      Frag bf;
      {
        int h = ht * 16 + (lane & 15);
        int k = k0 + ((lane >> 4) << 3);
        const unsigned short* src = (k < 320)
            ? hf + ((size_t)h * 128 + b) * 320 + k
            : hb + ((size_t)h * 128 + b) * 320 + (k - 320);
        bf.q[0] = *(const v4u*)src;
        bf.q[1] = *(const v4u*)(src + 16);
      }
      cc = wmma_bf(af, bf, cc);
    }
    for (int r = 0; r < 8; r++) {
      int p = pt * 16 + r + ((lane >> 4) << 3);
      int h = ht * 16 + (lane & 15);
      Exs[p * 64 + h] = __expf(cc[r]);
    }
  }
  __syncthreads();

  if (tid < 64) {
    float s = 0.f;
    for (int h = 0; h < 64; h++) s += Exs[tid * 64 + h] * mh[h];
    rs[tid] = s;
  } else if (tid < 128) {
    int h = tid - 64; float s = 0.f;
    for (int p = 0; p < 64; p++) s += Exs[p * 64 + h] * mp[p];
    cs[h] = s;
  }
  __syncthreads();

  for (int i = tid; i < 4096; i += 256) {
    int p = i >> 6, h = i & 63;
    float e = Exs[i];
    Alp[i]          = f2bf(e * mh[h] / rs[p]);
    Bet[h * 64 + p] = f2bf(e * mp[p] / cs[h]);
  }
  __syncthreads();

  // prem_attn = alpha @ Hy ; write m_a
  for (int ti = wave; ti < 160; ti += 8) {
    int pt = ti / 40, dt = ti % 40;
    v8f cc = v8f{};
    for (int kc = 0; kc < 2; kc++) {
      int k0 = kc * 32;
      Frag af; ldA(af, Alp, 64, pt * 16, k0, lane);
      Frag bf; ldB(bf, Tb, 64, dt * 16, k0, lane);
      cc = wmma_bf(af, bf, cc);
    }
    for (int r = 0; r < 8; r++) {
      int p = pt * 16 + r + ((lane >> 4) << 3);
      int d = dt * 16 + (lane & 15);
      float attn = cc[r];
      unsigned short prb = (d < 320) ? pf[((size_t)p * 128 + b) * 320 + d]
                                     : pb[((size_t)p * 128 + b) * 320 + d - 320];
      float pr = bf2f(prb);
      size_t base = ((size_t)p * 128 + b) * 2560 + d;
      ma[base]        = prb;
      ma[base + 640]  = f2bf(attn);
      ma[base + 1280] = f2bf(pr - attn);
      ma[base + 1920] = f2bf(pr * attn);
    }
  }
  __syncthreads();

  // PrT[d][p] overwrites Tb
  for (int i = tid; i < 640 * 64; i += 256) {
    int p = i / 640, d = i % 640;
    unsigned short v = (d < 320) ? pf[((size_t)p * 128 + b) * 320 + d]
                                 : pb[((size_t)p * 128 + b) * 320 + d - 320];
    Tb[d * 64 + p] = v;
  }
  __syncthreads();

  // hyp_attn = beta^T @ Pr ; write m_b
  for (int ti = wave; ti < 160; ti += 8) {
    int ht = ti / 40, dt = ti % 40;
    v8f cc = v8f{};
    for (int kc = 0; kc < 2; kc++) {
      int k0 = kc * 32;
      Frag af; ldA(af, Bet, 64, ht * 16, k0, lane);
      Frag bf; ldB(bf, Tb, 64, dt * 16, k0, lane);
      cc = wmma_bf(af, bf, cc);
    }
    for (int r = 0; r < 8; r++) {
      int h = ht * 16 + r + ((lane >> 4) << 3);
      int d = dt * 16 + (lane & 15);
      float attn = cc[r];
      unsigned short hyb = (d < 320) ? hf[((size_t)h * 128 + b) * 320 + d]
                                     : hb[((size_t)h * 128 + b) * 320 + d - 320];
      float hy = bf2f(hyb);
      size_t base = ((size_t)h * 128 + b) * 2560 + d;
      mb_[base]        = hyb;
      mb_[base + 640]  = f2bf(attn);
      mb_[base + 1280] = f2bf(hy - attn);
      mb_[base + 1920] = f2bf(hy * attn);
    }
  }
}

// ---------------------------------------------------------------------------
// Pooling: v[b, 2400] = [mean v1, mean v2, max v1, max v2] over t.
// ---------------------------------------------------------------------------
__global__ void k_pool(const float* v1f, const float* v1b,
                       const float* v2f, const float* v2b, float* v) {
  int b = blockIdx.x;
  for (int j = threadIdx.x; j < 600; j += blockDim.x) {
    int jj = (j < 300) ? j : (j - 300);
    const float* s1 = (j < 300) ? v1f : v1b;
    const float* s2 = (j < 300) ? v2f : v2b;
    float sum1 = 0.f, mx1 = -3.402823466e38f;
    float sum2 = 0.f, mx2 = -3.402823466e38f;
    for (int t = 0; t < 64; t++) {
      float x1 = s1[((size_t)t * 128 + b) * 320 + jj];
      float x2 = s2[((size_t)t * 128 + b) * 320 + jj];
      sum1 += x1; mx1 = fmaxf(mx1, x1);
      sum2 += x2; mx2 = fmaxf(mx2, x2);
    }
    v[(size_t)b * 2400 + j]        = sum1 * (1.f / 64.f);
    v[(size_t)b * 2400 + 600 + j]  = sum2 * (1.f / 64.f);
    v[(size_t)b * 2400 + 1200 + j] = mx1;
    v[(size_t)b * 2400 + 1800 + j] = mx2;
  }
}

// ---------------------------------------------------------------------------
// MLP head + classifier (fp32 for full output precision).
// ---------------------------------------------------------------------------
__global__ __launch_bounds__(512)
void k_head(const float* v, const float* mlpW, const float* mlpb,
            const float* clW, const float* clb, float* out) {
  __shared__ float hm[300];
  int b = blockIdx.x, j = threadIdx.x;
  if (j < 300) {
    float acc = mlpb[j];
    const float* w  = mlpW + (size_t)j * 2400;
    const float* vb = v + (size_t)b * 2400;
    for (int k = 0; k < 2400; k++) acc += vb[k] * w[k];
    hm[j] = tanhf(acc);
  }
  __syncthreads();
  if (j < 3) {
    float acc = clb[j];
    for (int k = 0; k < 300; k++) acc += hm[k] * clW[j * 300 + k];
    out[b * 3 + j] = acc;
  }
}

// ===========================================================================
// Host orchestration
// ===========================================================================
extern "C" void kernel_launch(void* const* d_in, const int* in_sizes, int n_in,
                              void* d_out, int out_size, void* d_ws, size_t ws_size,
                              hipStream_t stream) {
  (void)in_sizes; (void)n_in; (void)out_size; (void)ws_size;
  const int* premise    = (const int*)d_in[0];
  const int* hypothesis = (const int*)d_in[1];
  const float* embed_W  = (const float*)d_in[2];
  // prem_p 3..10, hyp_p 11..18, v1_p 19..26, v2_p 27..34 (each: Wih_f, Whh_f,
  // bih_f, bhh_f, Wih_b, Whh_b, bih_b, bhh_b)
  const float* mlpW = (const float*)d_in[35];
  const float* mlpb = (const float*)d_in[36];
  const float* clW  = (const float*)d_in[37];
  const float* clb  = (const float*)d_in[38];

  char* w = (char*)d_ws;
  size_t off = 0;
  auto bump = [&](size_t bytes) -> char* {
    char* p = w + off;
    off = (off + bytes + 255) & ~(size_t)255;
    return p;
  };

  unsigned short* Xp = (unsigned short*)bump((size_t)8192 * 320 * 2);
  unsigned short* Xh = (unsigned short*)bump((size_t)8192 * 320 * 2);
  unsigned short *W1ih[4], *W1hh[4], *W2ih[4], *W2hh[4], *hs1[4];
  float *b1[4], *b2[4], *hs2[4], *Zin[4];
  for (int c = 0; c < 4; c++) {
    W1ih[c] = (unsigned short*)bump((size_t)1280 * 320 * 2);
    W1hh[c] = (unsigned short*)bump((size_t)1280 * 320 * 2);
    b1[c]   = (float*)bump(1280 * 4);
  }
  for (int c = 0; c < 4; c++) {
    W2ih[c] = (unsigned short*)bump((size_t)1280 * 2560 * 2);
    W2hh[c] = (unsigned short*)bump((size_t)1280 * 320 * 2);
    b2[c]   = (float*)bump(1280 * 4);
  }
  for (int c = 0; c < 4; c++) hs1[c] = (unsigned short*)bump((size_t)64 * 128 * 320 * 2);
  for (int c = 0; c < 4; c++) hs2[c] = (float*)bump((size_t)64 * 128 * 320 * 4);
  unsigned short* ma  = (unsigned short*)bump((size_t)8192 * 2560 * 2);
  unsigned short* mb_ = (unsigned short*)bump((size_t)8192 * 2560 * 2);
  for (int c = 0; c < 4; c++) Zin[c] = (float*)bump((size_t)64 * 128 * 1280 * 4);
  float* vpool = (float*)bump((size_t)128 * 2400 * 4);

  // --- weight prep: layer1 chains {prem_f, prem_b, hyp_f, hyp_b} ---
  const int l1base[4][2] = {{3, 0}, {3, 4}, {11, 0}, {11, 4}};  // dict base + f/b offset
  for (int c = 0; c < 4; c++) {
    int i0 = l1base[c][0] + l1base[c][1];
    k_prep<<<1280, 256, 0, stream>>>((const float*)d_in[i0],
                                     (const float*)d_in[i0 + 1],
                                     (const float*)d_in[i0 + 2],
                                     (const float*)d_in[i0 + 3],
                                     300, 320, W1ih[c], W1hh[c], b1[c]);
  }
  const int l2base[4][2] = {{19, 0}, {19, 4}, {27, 0}, {27, 4}};
  for (int c = 0; c < 4; c++) {
    int i0 = l2base[c][0] + l2base[c][1];
    k_prep<<<1280, 256, 0, stream>>>((const float*)d_in[i0],
                                     (const float*)d_in[i0 + 1],
                                     (const float*)d_in[i0 + 2],
                                     (const float*)d_in[i0 + 3],
                                     2400, 2560, W2ih[c], W2hh[c], b2[c]);
  }

  // --- embeddings ---
  k_embed<<<8192, 256, 0, stream>>>(premise, embed_W, Xp);
  k_embed<<<8192, 256, 0, stream>>>(hypothesis, embed_W, Xh);

  // --- layer-1 input GEMMs (all timesteps in parallel) ---
  GemmP g1;
  g1.X[0] = Xp; g1.X[1] = Xp; g1.X[2] = Xh; g1.X[3] = Xh;
  for (int c = 0; c < 4; c++) { g1.W[c] = W1ih[c]; g1.bias[c] = b1[c]; g1.Z[c] = Zin[c]; }
  g1.K = 320;
  k_gemm<<<dim3(64, 10, 4), 256, 0, stream>>>(g1);

  // --- layer-1 recurrences (4 persistent blocks) ---
  LstmP l1;
  for (int c = 0; c < 4; c++) {
    l1.Zin[c] = Zin[c]; l1.Whh[c] = W1hh[c];
    l1.hs_bf[c] = hs1[c]; l1.hs_f[c] = nullptr; l1.rev[c] = c & 1;
  }
  k_lstm<<<4, 1024, 81920, stream>>>(l1);

  // --- cross attention + m_a/m_b ---
  k_attn<<<128, 256, 115712, stream>>>(premise, hypothesis,
                                       hs1[0], hs1[1], hs1[2], hs1[3], ma, mb_);

  // --- layer-2 input GEMMs (dominant FLOPs) ---
  GemmP g2;
  g2.X[0] = ma; g2.X[1] = ma; g2.X[2] = mb_; g2.X[3] = mb_;
  for (int c = 0; c < 4; c++) { g2.W[c] = W2ih[c]; g2.bias[c] = b2[c]; g2.Z[c] = Zin[c]; }
  g2.K = 2560;
  k_gemm<<<dim3(64, 10, 4), 256, 0, stream>>>(g2);

  // --- layer-2 recurrences ---
  LstmP l2;
  for (int c = 0; c < 4; c++) {
    l2.Zin[c] = Zin[c]; l2.Whh[c] = W2hh[c];
    l2.hs_bf[c] = nullptr; l2.hs_f[c] = hs2[c]; l2.rev[c] = c & 1;
  }
  k_lstm<<<4, 1024, 81920, stream>>>(l2);

  // --- pooling + head ---
  k_pool<<<128, 256, 0, stream>>>(hs2[0], hs2[1], hs2[2], hs2[3], vpool);
  k_head<<<128, 512, 0, stream>>>(vpool, mlpW, mlpb, clW, clb, (float*)d_out);
}
